// NeuromorphicLM_53369263620670
// MI455X (gfx1250) — matrix operand order repaired and checked by reading, not compiled, not run on hardware
//
#include <hip/hip_runtime.h>
#include <math.h>

typedef __bf16 bf16_t;
typedef __attribute__((ext_vector_type(16))) __bf16 v16bf;
typedef __attribute__((ext_vector_type(8)))  __bf16 v8bf;
typedef __attribute__((ext_vector_type(8)))  float  v8f;

#define BSZ   8
#define NSEQ  1024
#define DDIM  1024
#define NBANK 8
#define MSLOT 256

__device__ __forceinline__ v8f wmma_bf16(v16bf a, v16bf b, v8f c) {
  // D = A(16x32 bf16) * B(32x16 bf16) + C(16x16 f32)
  return __builtin_amdgcn_wmma_f32_16x16x32_bf16(false, a, false, b, (short)0, c,
                                                 false, false);
}

// ---------------------------------------------------------------------------
// Small reductions: bias_sum[d] = sum_b pm_bias[b,d];  lr_sum = sum(lr_pm)
// ---------------------------------------------------------------------------
__global__ void nlm_scalars(const float* __restrict__ pm_bias,
                            const float* __restrict__ lr_pm,
                            float* __restrict__ bias_sum,
                            float* __restrict__ lr_sum) {
  int d = threadIdx.x;
  if (d < DDIM) {
    float s = 0.f;
    #pragma unroll
    for (int b = 0; b < NBANK; ++b) s += pm_bias[b * DDIM + d];
    bias_sum[d] = s;
  }
  if (d == 0) {
    float s = 0.f;
    #pragma unroll
    for (int b = 0; b < NBANK; ++b) s += lr_pm[b];
    *lr_sum = s;
  }
}

// ---------------------------------------------------------------------------
// Row L2 norms (one block per row of length DDIM)
// ---------------------------------------------------------------------------
__global__ void nlm_rownorm(const float* __restrict__ src,
                            float* __restrict__ out) {
  __shared__ float red[256];
  size_t row = blockIdx.x;
  const float* p = src + row * (size_t)DDIM;
  float acc = 0.f;
  for (int i = threadIdx.x; i < DDIM; i += 256) { float v = p[i]; acc += v * v; }
  red[threadIdx.x] = acc;
  __syncthreads();
  for (int s = 128; s > 0; s >>= 1) {
    if (threadIdx.x < s) red[threadIdx.x] += red[threadIdx.x + s];
    __syncthreads();
  }
  if (threadIdx.x == 0) out[row] = sqrtf(red[0]);
}

// ---------------------------------------------------------------------------
// w_nov = sigmoid(H @ W_nov_w + b)   (tiny GEMM, N-cols = 8)
// ---------------------------------------------------------------------------
__global__ void nlm_novgate(const float* __restrict__ H,
                            const float* __restrict__ Wn,
                            const float* __restrict__ bn,
                            float* __restrict__ wnov) {
  int tid = blockIdx.x * 256 + threadIdx.x;  // [BSZ*NSEQ*NBANK)
  int b = tid & 7;
  int sn = tid >> 3;
  const float* h = H + (size_t)sn * DDIM;
  float acc = 0.f;
  for (int d = 0; d < DDIM; ++d) acc += h[d] * Wn[d * NBANK + b];
  acc += bn[b];
  wnov[tid] = 1.f / (1.f + __expf(-acc));
}

// ---------------------------------------------------------------------------
// Pack [s,n,d] f32 rows into WMMA A-operand bf16 lane layout.
// Tile order: [s][ntile(64)][ktile(32)][lane(32)][e(16)]
// A k order packed here (contiguous, same as B packers; not AMD's table order): K = kbase + e, kbase = (lane>=16)?16:0
// ---------------------------------------------------------------------------
__global__ void nlm_packA(const float* __restrict__ src,
                          const float* __restrict__ norm,  // may be null
                          bf16_t* __restrict__ dst) {
  int tid = blockIdx.x * 256 + threadIdx.x;   // [8*64*32*32)
  int lane = tid & 31; int t = tid >> 5;
  int kt = t & 31; t >>= 5;
  int nt = t & 63; int s = t >> 6;
  int l16 = lane & 15, half = lane >> 4;
  int n = nt * 16 + l16;
  size_t rowbase = ((size_t)s * NSEQ + n) * DDIM;
  float inv = 1.f;
  if (norm) inv = 1.f / (norm[s * NSEQ + n] + 1e-6f);
  int kb = kt * 32 + half * 16;
  bf16_t* o = dst + (size_t)tid * 16;
  #pragma unroll
  for (int e = 0; e < 16; ++e) {
    int k = kb + e;
    o[e] = (bf16_t)(src[rowbase + k] * inv);
  }
}

// ---------------------------------------------------------------------------
// Pack mem[s,b,m,d] into WMMA B-operand layouts over K=d (raw + normalized).
// Tile order: [sb(64)][mtile(16)][ktile(32)][lane(32)][e(16)]
// B k order packed here (contiguous, same as packA; not AMD's table order): N = lane%16, K = (lane>=16?16:0) + e
// ---------------------------------------------------------------------------
__global__ void nlm_packB(const float* __restrict__ mem,
                          const float* __restrict__ mnorm,
                          bf16_t* __restrict__ rawB,
                          bf16_t* __restrict__ nrmB) {
  int tid = blockIdx.x * 256 + threadIdx.x;   // [64*16*32*32)
  int lane = tid & 31; int t = tid >> 5;
  int kt = t & 31; t >>= 5;
  int mt = t & 15; int sb = t >> 4;
  int l16 = lane & 15, half = lane >> 4;
  int m = mt * 16 + l16;
  size_t rowbase = ((size_t)sb * MSLOT + m) * DDIM;
  float inv = 1.f / (mnorm[sb * MSLOT + m] + 1e-6f);
  int d0 = kt * 32 + half * 16;
  bf16_t* o1 = rawB + (size_t)tid * 16;
  bf16_t* o2 = nrmB + (size_t)tid * 16;
  #pragma unroll
  for (int e = 0; e < 16; ++e) {
    float v = mem[rowbase + d0 + e];
    o1[e] = (bf16_t)v;
    o2[e] = (bf16_t)(v * inv);
  }
}

// ---------------------------------------------------------------------------
// Pack mem[s,b,m,d] into B-operand layout over K=m, N=d (for attn @ mem).
// Tile order: [sb(64)][dtile(64)][ktile(8)][lane(32)][e(16)]
// ---------------------------------------------------------------------------
__global__ void nlm_packB2(const float* __restrict__ mem,
                           bf16_t* __restrict__ dst) {
  int tid = blockIdx.x * 256 + threadIdx.x;   // [64*64*8*32)
  int lane = tid & 31; int t = tid >> 5;
  int kt = t & 7; t >>= 3;
  int dt = t & 63; int sb = t >> 6;
  int l16 = lane & 15, half = lane >> 4;
  int d = dt * 16 + l16;
  int m0 = kt * 32 + half * 16;
  const float* base = mem + (size_t)sb * MSLOT * DDIM + d;
  bf16_t* o = dst + (size_t)tid * 16;
  #pragma unroll
  for (int e = 0; e < 16; ++e)
    o[e] = (bf16_t)base[(size_t)(m0 + e) * DDIM];
}

// ---------------------------------------------------------------------------
// Fused attention kernel: one block = (s, bank, 32 query rows), 2 waves,
// each wave owns a 16-row strip.
//   Pass 1: sim = wq @ mk^T, fused max over all 256 slots -> novelty
//   Pass 2: scores = seed @ mem^T * dscale -> LDS -> softmax (exp only)
//   Pass 3: em = exp_scores @ mem, scaled by 1/rowsum, atomic bank-sum
// ---------------------------------------------------------------------------
__global__ void __launch_bounds__(64) nlm_attn(
    const bf16_t* __restrict__ seedA, const bf16_t* __restrict__ wqA,
    const bf16_t* __restrict__ memB,  const bf16_t* __restrict__ mkB,
    const bf16_t* __restrict__ memB2,
    const float* __restrict__ wnov,   float* __restrict__ novelty,
    float* __restrict__ em_acc) {
  __shared__ __align__(16) float  s_scores[2][16][256];  // 32 KB
  __shared__ __align__(16) bf16_t s_attn[2][16][256];    // 16 KB
  __shared__ float s_linv[2][16];

  const int wave = threadIdx.x >> 5;
  const int lane = threadIdx.x & 31;
  const int l16  = lane & 15;
  const int half = lane >> 4;
  const int s = blockIdx.z, b = blockIdx.y;
  const int ntA = blockIdx.x * 2 + wave;   // 16-row tile index in [0,64)
  const int n0 = ntA * 16;
  const int sb = s * NBANK + b;
  const v8f vz = {0.f, 0.f, 0.f, 0.f, 0.f, 0.f, 0.f, 0.f};

  // ---- Pass 1: max cosine similarity ----
  const v16bf* aW = (const v16bf*)wqA + ((size_t)(s * 64 + ntA) * 32) * 32 + lane;
  float smax[8];
  #pragma unroll
  for (int r = 0; r < 8; ++r) smax[r] = -1e30f;
  for (int mt = 0; mt < 16; ++mt) {
    const v16bf* bK = (const v16bf*)mkB + ((size_t)(sb * 16 + mt) * 32) * 32 + lane;
    v8f acc = vz;
    for (int kt = 0; kt < 32; ++kt) {
      __builtin_prefetch((const void*)(bK + (size_t)(kt + 1) * 32), 0, 0);
      acc = wmma_bf16(aW[(size_t)kt * 32], bK[(size_t)kt * 32], acc);
    }
    #pragma unroll
    for (int r = 0; r < 8; ++r) smax[r] = fmaxf(smax[r], acc[r]);
  }
  #pragma unroll
  for (int r = 0; r < 8; ++r) {
    #pragma unroll
    for (int off = 1; off < 16; off <<= 1)
      smax[r] = fmaxf(smax[r], __shfl_xor(smax[r], off));
  }
  if (l16 == 0) {
    #pragma unroll
    for (int r = 0; r < 8; ++r) {
      int n = n0 + half * 8 + r;
      size_t gi = ((size_t)s * NSEQ + n) * NBANK + b;
      novelty[gi] = wnov[gi] * fmaxf(0.f, 1.f - smax[r]);
    }
  }

  // ---- Pass 2: attention scores ----
  const float dscale = 0.03125f;  // 1/sqrt(1024)
  const v16bf* aS = (const v16bf*)seedA + ((size_t)(s * 64 + ntA) * 32) * 32 + lane;
  for (int mt = 0; mt < 16; ++mt) {
    const v16bf* bK = (const v16bf*)memB + ((size_t)(sb * 16 + mt) * 32) * 32 + lane;
    v8f acc = vz;
    for (int kt = 0; kt < 32; ++kt)
      acc = wmma_bf16(aS[(size_t)kt * 32], bK[(size_t)kt * 32], acc);
    #pragma unroll
    for (int r = 0; r < 8; ++r)
      s_scores[wave][half * 8 + r][mt * 16 + l16] = acc[r] * dscale;
  }
  __syncthreads();

  // ---- softmax over m (2 lanes per row; unnormalized exp, keep 1/sum) ----
  {
    int row = l16;
    int c0 = half * 128;
    float rmax = -1e30f;
    for (int c = 0; c < 128; ++c) rmax = fmaxf(rmax, s_scores[wave][row][c0 + c]);
    rmax = fmaxf(rmax, __shfl_xor(rmax, 16));
    float sum = 0.f;
    for (int c = 0; c < 128; ++c) {
      float e = __expf(s_scores[wave][row][c0 + c] - rmax);
      s_attn[wave][row][c0 + c] = (bf16_t)e;
      sum += e;
    }
    sum += __shfl_xor(sum, 16);
    if (half == 0) s_linv[wave][row] = 1.f / sum;
  }
  __syncthreads();

  // ---- Pass 3: em_read = exp_scores @ mem, bank-summed via f32 atomics ----
  const int kbA = half ? 16 : 0;
  const v8bf* arow = (const v8bf*)&s_attn[wave][l16][0];
  float linv[8];
  #pragma unroll
  for (int r = 0; r < 8; ++r) linv[r] = s_linv[wave][half * 8 + r];

  for (int dt = 0; dt < 64; ++dt) {
    const v16bf* bD = (const v16bf*)memB2 + ((size_t)(sb * 64 + dt) * 8) * 32 + lane;
    v8f acc = vz;
    #pragma unroll
    for (int kt = 0; kt < 8; ++kt) {
      v8bf c0 = arow[(kt * 32 + kbA) >> 3];
      v8bf c1 = arow[(kt * 32 + kbA + 8) >> 3];
      v16bf a = __builtin_shufflevector(c0, c1, 0, 1, 2, 3, 4, 5, 6, 7,
                                        8, 9, 10, 11, 12, 13, 14, 15);
      acc = wmma_bf16(a, bD[(size_t)kt * 32], acc);
    }
    #pragma unroll
    for (int r = 0; r < 8; ++r) {
      int n = n0 + half * 8 + r;
      int d = dt * 16 + l16;
      atomicAdd(&em_acc[((size_t)s * NSEQ + n) * DDIM + d], acc[r] * linv[r]);
    }
  }
}

// ---------------------------------------------------------------------------
// nov_sum[s,n] = sum_b novelty[s,n,b]
// ---------------------------------------------------------------------------
__global__ void nlm_novsum(const float* __restrict__ novelty,
                           float* __restrict__ novsum) {
  int tid = blockIdx.x * 256 + threadIdx.x;  // [BSZ*NSEQ)
  float acc = 0.f;
  #pragma unroll
  for (int b = 0; b < NBANK; ++b) acc += novelty[(size_t)tid * NBANK + b];
  novsum[tid] = acc;
}

// ---------------------------------------------------------------------------
// Causal scans + final combine (coalesced over d, sequential over n):
// out = H + H*(bias_sum + lr_sum*cumsum(surprise)) + em + cumsum(nov_sum*wc)
// ---------------------------------------------------------------------------
__global__ void nlm_scan(const float* __restrict__ H,
                         const float* __restrict__ surprise,
                         const float* __restrict__ wc,
                         const float* __restrict__ em,
                         const float* __restrict__ novsum,
                         const float* __restrict__ bias_sum,
                         const float* __restrict__ lr_sum,
                         float* __restrict__ out) {
  int d = blockIdx.x * 256 + threadIdx.x;
  int s = blockIdx.y;
  float bias = bias_sum[d];
  float lr = *lr_sum;
  float cs = 0.f, ce = 0.f;
  size_t idx = (size_t)s * NSEQ * DDIM + d;
  const float* nv = novsum + (size_t)s * NSEQ;
  for (int n = 0; n < NSEQ; ++n, idx += DDIM) {
    float sur = surprise[idx];
    float h = H[idx];
    float w = wc[idx];
    cs += sur;
    ce += nv[n] * w;
    out[idx] = h + h * (bias + lr * cs) + em[idx] + ce;
  }
}

// ---------------------------------------------------------------------------
extern "C" void kernel_launch(void* const* d_in, const int* in_sizes, int n_in,
                              void* d_out, int out_size, void* d_ws, size_t ws_size,
                              hipStream_t stream) {
  const float* H        = (const float*)d_in[0];
  const float* seed     = (const float*)d_in[1];
  const float* w_cand   = (const float*)d_in[2];
  const float* surprise = (const float*)d_in[3];
  const float* W_nov_w  = (const float*)d_in[4];
  const float* W_nov_b  = (const float*)d_in[5];
  const float* pm_bias  = (const float*)d_in[6];
  const float* lr_pm    = (const float*)d_in[7];
  const float* mem      = (const float*)d_in[8];
  float* out = (float*)d_out;

  char* ws = (char*)d_ws;
  size_t off = 0;
  auto take = [&](size_t bytes) -> void* {
    size_t o = (off + 255) & ~(size_t)255;
    off = o + bytes;
    return (void*)(ws + o);
  };

  float*  w_lr      = (float*)take(4);
  float*  w_bias    = (float*)take((size_t)DDIM * 4);
  float*  w_wnorm   = (float*)take((size_t)BSZ * NSEQ * 4);
  float*  w_memnorm = (float*)take((size_t)BSZ * NBANK * MSLOT * 4);
  float*  w_wnov    = (float*)take((size_t)BSZ * NSEQ * NBANK * 4);
  float*  w_novel   = (float*)take((size_t)BSZ * NSEQ * NBANK * 4);
  float*  w_novsum  = (float*)take((size_t)BSZ * NSEQ * 4);
  bf16_t* p_seedA   = (bf16_t*)take((size_t)BSZ * NSEQ * DDIM * 2);
  bf16_t* p_wqA     = (bf16_t*)take((size_t)BSZ * NSEQ * DDIM * 2);
  bf16_t* p_memB    = (bf16_t*)take((size_t)BSZ * NBANK * MSLOT * DDIM * 2);
  bf16_t* p_mkB     = (bf16_t*)take((size_t)BSZ * NBANK * MSLOT * DDIM * 2);
  bf16_t* p_memB2   = (bf16_t*)take((size_t)BSZ * NBANK * MSLOT * DDIM * 2);
  float*  w_emacc   = (float*)take((size_t)BSZ * NSEQ * DDIM * 4);

  hipMemsetAsync(w_emacc, 0, (size_t)BSZ * NSEQ * DDIM * 4, stream);

  nlm_scalars<<<1, 1024, 0, stream>>>(pm_bias, lr_pm, w_bias, w_lr);
  nlm_rownorm<<<BSZ * NSEQ, 256, 0, stream>>>(w_cand, w_wnorm);
  nlm_rownorm<<<BSZ * NBANK * MSLOT, 256, 0, stream>>>(mem, w_memnorm);
  nlm_novgate<<<(BSZ * NSEQ * NBANK) / 256, 256, 0, stream>>>(H, W_nov_w, W_nov_b, w_wnov);

  nlm_packA<<<(BSZ * 64 * 32 * 32) / 256, 256, 0, stream>>>(seed, nullptr, p_seedA);
  nlm_packA<<<(BSZ * 64 * 32 * 32) / 256, 256, 0, stream>>>(w_cand, w_wnorm, p_wqA);
  nlm_packB<<<(BSZ * NBANK * 16 * 32 * 32) / 256, 256, 0, stream>>>(mem, w_memnorm, p_memB, p_mkB);
  nlm_packB2<<<(BSZ * NBANK * 64 * 8 * 32) / 256, 256, 0, stream>>>(mem, p_memB2);

  nlm_attn<<<dim3(NSEQ / 32, NBANK, BSZ), 64, 0, stream>>>(
      p_seedA, p_wqA, p_memB, p_mkB, p_memB2, w_wnov, w_novel, w_emacc);

  nlm_novsum<<<(BSZ * NSEQ) / 256, 256, 0, stream>>>(w_novel, w_novsum);
  nlm_scan<<<dim3(DDIM / 256, BSZ), 256, 0, stream>>>(
      H, surprise, w_cand, w_emacc, w_novsum, w_bias, w_lr, out);
}
